// CausalSelfAttention_30176440222065
// MI455X (gfx1250) — compile-verified, matrix operand order checked by reading, not run on hardware
//
#include <hip/hip_runtime.h>
#include <hip/hip_bf16.h>

typedef __attribute__((ext_vector_type(16))) _Float16 v16h;
typedef __attribute__((ext_vector_type(8)))  float    v8f;
typedef __attribute__((ext_vector_type(4)))  int      v4i;

// ---- optional CDNA5 async global->LDS path (guarded; falls back cleanly) ----
#if defined(__has_builtin)
#  if __has_builtin(__builtin_amdgcn_global_load_async_to_lds_b128)
#    define HAVE_ASYNC_LDS 1
#  else
#    define HAVE_ASYNC_LDS 0
#  endif
#  if __has_builtin(__builtin_amdgcn_s_wait_asynccnt)
#    define HAVE_WAIT_ASYNC 1
#  else
#    define HAVE_WAIT_ASYNC 0
#  endif
#else
#  define HAVE_ASYNC_LDS 0
#  define HAVE_WAIT_ASYNC 0
#endif

#if HAVE_ASYNC_LDS
typedef v4i __attribute__((address_space(1)))* g_v4i_p;   // global int4*
typedef v4i __attribute__((address_space(3)))* l_v4i_p;   // LDS int4*
#endif

// ---------------------------------------------------------------------------
// GEMM with bias: Out[M,N] = A[M,K] @ B[K,N] + bias[N]
// Double-buffered LDS staging; f16 WMMA with f32 accumulate.
// Tiles: BM=128, BN=128, BK=32. 256 threads = 8 waves, wave tile 32x64.
// ---------------------------------------------------------------------------
template<bool A_F32, bool OUT_F16>
__global__ __launch_bounds__(256)
void gemm_bias_wmma(const void* __restrict__ Ap, const float* __restrict__ Bp,
                    const float* __restrict__ bias, void* __restrict__ Op,
                    int M, int N, int K)
{
  constexpr int BM = 128, BN = 128, BK = 32;
  __shared__ _Float16 As[2][BM][BK + 8];
  __shared__ _Float16 Bs[2][BK][BN + 8];

  const int tid   = threadIdx.x;
  const int lane  = tid & 31;
  const int wave  = tid >> 5;
  const int lrow  = lane & 15;
  const int lhalf = lane >> 4;
  const int kbase = lhalf * 8;          // K pattern base per ISA A/B layout
  const int wm    = (wave >> 1) * 32;   // wave row offset: 0,32,64,96
  const int wn    = (wave & 1) * 64;    // wave col offset: 0,64
  const int bM    = blockIdx.y * BM;
  const int bN    = blockIdx.x * BN;

  auto stage = [&](int buf, int k0) {
    {
      const int r  = tid >> 1;
      const int kc = (tid & 1) * 16;
      if constexpr (A_F32) {
        const float* g = (const float*)Ap + (size_t)(bM + r) * K + k0 + kc;
        __builtin_prefetch(g + BK, 0, 1);
#pragma unroll
        for (int j = 0; j < 16; ++j) As[buf][r][kc + j] = (_Float16)g[j];
      } else {
        const _Float16* g = (const _Float16*)Ap + (size_t)(bM + r) * K + k0 + kc;
        __builtin_prefetch(g + BK, 0, 1);
#pragma unroll
        for (int j = 0; j < 16; ++j) As[buf][r][kc + j] = g[j];
      }
    }
    {
      const int r  = tid >> 3;
      const int nc = (tid & 7) * 16;
      const float* g = Bp + (size_t)(k0 + r) * N + bN + nc;
      __builtin_prefetch(g + (size_t)BK * N, 0, 1);
#pragma unroll
      for (int j = 0; j < 16; ++j) Bs[buf][r][nc + j] = (_Float16)g[j];
    }
  };

  v8f acc[2][4];
#pragma unroll
  for (int mi = 0; mi < 2; ++mi)
#pragma unroll
    for (int ni = 0; ni < 4; ++ni)
#pragma unroll
      for (int v = 0; v < 8; ++v) acc[mi][ni][v] = 0.0f;

  const int nk = K / BK;
  stage(0, 0);
  for (int kt = 0; kt < nk; ++kt) {
    __syncthreads();                    // staged tile kt is ready
    if (kt + 1 < nk) stage((kt + 1) & 1, (kt + 1) * BK);
    const int buf = kt & 1;

    v16h af[2], bf[4];
#pragma unroll
    for (int mi = 0; mi < 2; ++mi)
#pragma unroll
      for (int j = 0; j < 16; ++j) {
        const int kk = kbase + (j < 8 ? j : 8 + j);
        af[mi][j] = As[buf][wm + mi * 16 + lrow][kk];
      }
#pragma unroll
    for (int ni = 0; ni < 4; ++ni)
#pragma unroll
      for (int j = 0; j < 16; ++j) {
        const int kk = kbase + (j < 8 ? j : 8 + j);
        bf[ni][j] = Bs[buf][kk][wn + ni * 16 + lrow];
      }
#pragma unroll
    for (int mi = 0; mi < 2; ++mi)
#pragma unroll
      for (int ni = 0; ni < 4; ++ni)
        acc[mi][ni] = __builtin_amdgcn_wmma_f32_16x16x32_f16(
            false, af[mi], false, bf[ni], (short)0, acc[mi][ni], false, false);
  }

#pragma unroll
  for (int mi = 0; mi < 2; ++mi)
#pragma unroll
    for (int ni = 0; ni < 4; ++ni)
#pragma unroll
      for (int v = 0; v < 8; ++v) {
        const int row = bM + wm + mi * 16 + lhalf * 8 + v;
        const int col = bN + wn + ni * 16 + lrow;
        const float val = acc[mi][ni][v] + bias[col];
        if constexpr (OUT_F16)
          ((_Float16*)Op)[(size_t)row * N + col] = (_Float16)val;
        else
          ((float*)Op)[(size_t)row * N + col] = val;
      }
}

// ---------------------------------------------------------------------------
// Flash attention over f16 qkv workspace.
// qkv layout: [(b*T + t) * 3072 + {0:q,1024:k,2048:v} + h*64 + d]
// Grid: (T/64) * B * H blocks of 128 threads (4 waves). Wave owns 16 q rows.
// K/V tiles (32 keys x 64 d) staged cooperatively in LDS (async path when
// available). Online softmax; P bounced through LDS for C->A layout.
// ---------------------------------------------------------------------------
__global__ __launch_bounds__(128)
void attn_flash_wmma(const _Float16* __restrict__ qkv, _Float16* __restrict__ y)
{
  constexpr int T = 2048, H = 16, C3 = 3072, C = 1024;
  const int tile = blockIdx.x;
  const int qt = tile & 31;            // T/64 = 32 q tiles
  const int bh = tile >> 5;
  const int h  = bh & (H - 1);
  const int b  = bh >> 4;

  const int tid   = threadIdx.x;
  const int lane  = tid & 31;
  const int wave  = tid >> 5;
  const int lrow  = lane & 15;
  const int lhalf = lane >> 4;
  const int kbase = lhalf * 8;
  const int q0 = qt * 64 + wave * 16;

  const size_t base = (size_t)b * T * C3;
  const _Float16* Qp = qkv + base + (size_t)h * 64;
  const _Float16* Kp = qkv + base + 1024 + (size_t)h * 64;
  const _Float16* Vp = qkv + base + 2048 + (size_t)h * 64;

  __shared__ alignas(16) _Float16 Kt[32][72];  // 32 keys x 64 d (pad = 144B row, 16B aligned)
  __shared__ alignas(16) _Float16 Vt[32][72];
  __shared__ _Float16 Plds[4][16][40];

  // Q fragment: 16x64 = two 16x32 A-fragments, held for the whole pass
  v16h qf[2];
  {
    const int qrow = q0 + lrow;
#pragma unroll
    for (int c = 0; c < 2; ++c)
#pragma unroll
      for (int j = 0; j < 16; ++j) {
        const int d = c * 32 + kbase + (j < 8 ? j : 8 + j);
        qf[c][j] = Qp[(size_t)qrow * C3 + d];
      }
  }

  float mi_[8], li_[8];
  v8f acc[4];
#pragma unroll
  for (int v = 0; v < 8; ++v) { mi_[v] = -__builtin_inff(); li_[v] = 0.0f; }
#pragma unroll
  for (int dt = 0; dt < 4; ++dt)
#pragma unroll
    for (int v = 0; v < 8; ++v) acc[dt][v] = 0.0f;

  const int nkb = qt * 2 + 2;          // uniform across block (causal extent)
  for (int it = 0; it < nkb; ++it) {
    const int kb = it * 32;

    // ---- cooperative K/V tile staging: 32 rows x 64 halves each
#if HAVE_ASYNC_LDS
    {
#pragma unroll
      for (int u = 0; u < 2; ++u) {
        const int c = tid * 2 + u;           // 0..255 chunks of 8 halves (16B)
        const int r = c >> 3, seg = (c & 7) * 8;
        __builtin_amdgcn_global_load_async_to_lds_b128(
            (g_v4i_p)(Kp + (size_t)(kb + r) * C3 + seg),
            (l_v4i_p)&Kt[r][seg], 0, 0);
        __builtin_amdgcn_global_load_async_to_lds_b128(
            (g_v4i_p)(Vp + (size_t)(kb + r) * C3 + seg),
            (l_v4i_p)&Vt[r][seg], 0, 0);
      }
    }
#if HAVE_WAIT_ASYNC
    __builtin_amdgcn_s_wait_asynccnt(0);
#endif
#else
    {
      const int r  = tid >> 2;
      const int dc = (tid & 3) * 16;
      const _Float16* gk = Kp + (size_t)(kb + r) * C3 + dc;
      const _Float16* gv = Vp + (size_t)(kb + r) * C3 + dc;
#pragma unroll
      for (int j = 0; j < 16; ++j) Kt[r][dc + j] = gk[j];
#pragma unroll
      for (int j = 0; j < 16; ++j) Vt[r][dc + j] = gv[j];
    }
#endif
    __syncthreads();

    // ---- S = Q @ K^T for 32 keys (two 16-col tiles) from LDS
    v8f s[2];
#pragma unroll
    for (int t = 0; t < 2; ++t) {
#pragma unroll
      for (int v = 0; v < 8; ++v) s[t][v] = 0.0f;
      const int kl = t * 16 + lrow;        // key within tile
#pragma unroll
      for (int c = 0; c < 2; ++c) {
        v16h kf;
#pragma unroll
        for (int j = 0; j < 16; ++j) {
          const int d = c * 32 + kbase + (j < 8 ? j : 8 + j);
          kf[j] = Kt[kl][d];
        }
        s[t] = __builtin_amdgcn_wmma_f32_16x16x32_f16(
            false, qf[c], false, kf, (short)0, s[t], false, false);
      }
    }

    // ---- scale + causal mask (col = key, row = v + 8*lhalf)
#pragma unroll
    for (int t = 0; t < 2; ++t) {
      const int key = kb + t * 16 + lrow;
#pragma unroll
      for (int v = 0; v < 8; ++v) {
        const int r = lhalf * 8 + v;
        float val = s[t][v] * 0.125f;
        if (key > q0 + r) val = -__builtin_inff();
        s[t][v] = val;
      }
    }

    // ---- online softmax per row; stage P into LDS
#pragma unroll
    for (int v = 0; v < 8; ++v) {
      float tmax = fmaxf(s[0][v], s[1][v]);
#pragma unroll
      for (int off = 8; off >= 1; off >>= 1)
        tmax = fmaxf(tmax, __shfl_xor(tmax, off, 32));
      const float mnew = fmaxf(mi_[v], tmax);
      const float corr = __expf(mi_[v] - mnew);
      mi_[v] = mnew;
      const float p0 = __expf(s[0][v] - mnew);
      const float p1 = __expf(s[1][v] - mnew);
      float psum = p0 + p1;
#pragma unroll
      for (int off = 8; off >= 1; off >>= 1)
        psum += __shfl_xor(psum, off, 32);
      li_[v] = li_[v] * corr + psum;
#pragma unroll
      for (int dt = 0; dt < 4; ++dt) acc[dt][v] *= corr;
      const int r = lhalf * 8 + v;
      Plds[wave][r][lrow]      = (_Float16)p0;
      Plds[wave][r][16 + lrow] = (_Float16)p1;
    }
    __syncthreads();

    // ---- P (16x32) as A-fragment, V (32x64) from LDS as 4 B-fragments
    v16h pf;
#pragma unroll
    for (int j = 0; j < 16; ++j) {
      const int kk = kbase + (j < 8 ? j : 8 + j);
      pf[j] = Plds[wave][lrow][kk];
    }
#pragma unroll
    for (int dt = 0; dt < 4; ++dt) {
      v16h vf;
#pragma unroll
      for (int j = 0; j < 16; ++j) {
        const int kk = kbase + (j < 8 ? j : 8 + j);
        vf[j] = Vt[kk][dt * 16 + lrow];
      }
      acc[dt] = __builtin_amdgcn_wmma_f32_16x16x32_f16(
          false, pf, false, vf, (short)0, acc[dt], false, false);
    }
    __syncthreads();                     // protect Kt/Vt/Plds before restage
  }

  // ---- normalize and write y (merge heads back to [B,T,C])
#pragma unroll
  for (int dt = 0; dt < 4; ++dt)
#pragma unroll
    for (int v = 0; v < 8; ++v) {
      const int r = lhalf * 8 + v;
      const int q = q0 + r;
      const float o = acc[dt][v] / li_[v];
      y[(size_t)(b * T + q) * C + h * 64 + dt * 16 + lrow] = (_Float16)o;
    }
}

// ---------------------------------------------------------------------------
extern "C" void kernel_launch(void* const* d_in, const int* in_sizes, int n_in,
                              void* d_out, int out_size, void* d_ws, size_t ws_size,
                              hipStream_t stream) {
  const float* x      = (const float*)d_in[0];
  const float* w_attn = (const float*)d_in[1];
  const float* b_attn = (const float*)d_in[2];
  const float* w_proj = (const float*)d_in[3];
  const float* b_proj = (const float*)d_in[4];
  float* out = (float*)d_out;

  constexpr int B = 4, T = 2048, C = 1024, H = 16;
  constexpr int M = B * T;                      // 8192 rows

  _Float16* qkvw = (_Float16*)d_ws;                                   // [M, 3C] f16
  _Float16* yw   = (_Float16*)((char*)d_ws +
                    (size_t)M * 3 * C * sizeof(_Float16));            // [M, C] f16

  // 1) QKV projection: x(f32) @ w_attn + b_attn -> qkv (f16)
  gemm_bias_wmma<true, true><<<dim3((3 * C) / 128, M / 128), dim3(256), 0, stream>>>(
      x, w_attn, b_attn, qkvw, M, 3 * C, C);

  // 2) Fused causal flash attention -> y (f16)
  attn_flash_wmma<<<dim3((T / 64) * B * H), dim3(128), 0, stream>>>(qkvw, yw);

  // 3) Output projection: y(f16) @ w_proj + b_proj -> out (f32)
  gemm_bias_wmma<false, false><<<dim3(C / 128, M / 128), dim3(256), 0, stream>>>(
      yw, w_proj, b_proj, out, M, C, C);
}
